// RelativeAttention_24532853194791
// MI455X (gfx1250) — matrix operand order repaired and checked by reading, not compiled, not run on hardware
//
#include <hip/hip_runtime.h>
#include <hip/hip_bf16.h>
#include <math.h>

// Problem constants (shapes fixed by the harness/reference)
#define D_DIM   384
#define A_DIM   1024
#define H_DIM   512
#define TILE_M  64       // rows of x per block (4 x 16-row WMMA sub-tiles)
#define NTHREADS 512     // 16 wave32
#define XS      392      // padded LDS row stride (f16) for xn  (16B-multiple, bank-spread)
#define SWS     1032     // padded LDS row stride (f16) for sims->weights (in-place)
#define INV_SQRT_D 0.051031036307982884f   // 1/sqrt(384)
#define EPS     1e-12f
#define BIN     0.05f
#define INV_BIN 20.0f

typedef __attribute__((ext_vector_type(16))) _Float16 v16h;
typedef __attribute__((ext_vector_type(8)))  _Float16 v8h;
typedef __attribute__((ext_vector_type(8)))  float    v8f;

union AFrag { v16h v; v8h h8[2]; };

// ---------------------------------------------------------------------------
// Prep 1: L2-normalize anchors, fold in 1/sqrt(D), emit f16 [A][D]
// ---------------------------------------------------------------------------
__global__ void ra_prep_anchors(const float* __restrict__ anchors,
                                _Float16* __restrict__ ah) {
    const int a   = blockIdx.x;     // 0..A_DIM-1
    const int tid = threadIdx.x;    // 128 threads
    const float* ar = anchors + (size_t)a * D_DIM;

    float ss = 0.0f;
    for (int d = tid; d < D_DIM; d += 128) { float v = ar[d]; ss += v * v; }

    __shared__ float red[128];
    red[tid] = ss;
    __syncthreads();
    for (int s = 64; s > 0; s >>= 1) {
        if (tid < s) red[tid] += red[tid + s];
        __syncthreads();
    }
    const float nrm = sqrtf(red[0]);
    const float inv = INV_SQRT_D / fmaxf(nrm, EPS);

    for (int d = tid; d < D_DIM; d += 128)
        ah[(size_t)a * D_DIM + d] = (_Float16)(ar[d] * inv);
}

// ---------------------------------------------------------------------------
// Prep 2: transpose values [A][H] f32 -> vt [H][A] f16
// ---------------------------------------------------------------------------
__global__ void ra_prep_values(const float* __restrict__ values,
                               _Float16* __restrict__ vt) {
    const int idx = blockIdx.x * 256 + threadIdx.x;   // 0 .. H*A-1
    const int h = idx >> 10;          // / A_DIM
    const int a = idx & (A_DIM - 1);
    vt[idx] = (_Float16)values[(size_t)a * H_DIM + h];
}

// ---------------------------------------------------------------------------
// Main fused kernel: one block = 64 rows of x, 512 threads = 16 wave32.
// Each wave computes FOUR 16-row M sub-tiles so every B fragment feeds 4 WMMAs
// (quarters the L2-resident anchor/value re-read traffic per output row).
// qsims goes to global f32-exact from fragments; sims live in LDS as f16 only
// for the softmax and are overwritten in place by the f16 weights.
// ---------------------------------------------------------------------------
__global__ void __launch_bounds__(NTHREADS)
ra_main(const float* __restrict__ x,
        const _Float16* __restrict__ ah,     // [A][D] f16, scale folded in
        const _Float16* __restrict__ vt,     // [H][A] f16
        float* __restrict__ out,             // [B][H]
        float* __restrict__ qsims) {         // [B][A]
    __shared__ __align__(32) _Float16 xnbuf[TILE_M * XS];    //  50 KB
    __shared__ __align__(32) _Float16 swbuf[TILE_M * SWS];   // 132 KB (sims -> weights)

    const int tid     = threadIdx.x;
    const int lane    = tid & 31;
    const int wv      = tid >> 5;          // wave id 0..15
    const int mrow    = lane & 15;         // fragment row / column index
    const int khalf   = lane >> 4;         // 0 or 1
    const int rowbase = blockIdx.x * TILE_M;

    // ---------------- Stage 0: normalize x rows -> LDS f16 ----------------
    {
        const int row = tid >> 3;          // 0..63
        const int sub = tid & 7;           // 8 threads per row
        const float* xr = x + (size_t)(rowbase + row) * D_DIM;
        float ss = 0.0f;
        for (int d = sub; d < D_DIM; d += 8) { float v = xr[d]; ss += v * v; }
        ss += __shfl_xor(ss, 1, 32);
        ss += __shfl_xor(ss, 2, 32);
        ss += __shfl_xor(ss, 4, 32);
        const float inv = 1.0f / fmaxf(sqrtf(ss), EPS);
        for (int d = sub; d < D_DIM; d += 8)
            xnbuf[row * XS + d] = (_Float16)(xr[d] * inv);
    }
    __syncthreads();

    // ---------------- Stage 1: sims via WMMA (M=64, N=1024, K=384) --------
    // wave wv owns anchor columns [wv*64, wv*64+64) as 4 N-tiles x 4 M-subs
    v8f acc[4][4];
#pragma unroll
    for (int ms = 0; ms < 4; ++ms)
#pragma unroll
        for (int t = 0; t < 4; ++t) acc[ms][t] = (v8f)0.0f;

    for (int kt = 0; kt < D_DIM / 32; ++kt) {      // 12 k-steps of 32
        const int k0 = kt * 32 + khalf * 16;
        AFrag a[4];
#pragma unroll
        for (int ms = 0; ms < 4; ++ms) {
            a[ms].h8[0] = *(const v8h*)&xnbuf[(ms * 16 + mrow) * XS + k0];
            a[ms].h8[1] = *(const v8h*)&xnbuf[(ms * 16 + mrow) * XS + k0 + 8];
        }
#pragma unroll
        for (int t = 0; t < 4; ++t) {
            const int arow = wv * 64 + t * 16 + mrow;   // anchor index = column n
            const _Float16* bp = ah + (size_t)arow * D_DIM + kt * 32 + khalf * 16;
            const v16h b = *(const v16h*)bp;
#pragma unroll
            for (int ms = 0; ms < 4; ++ms)
                acc[ms][t] = __builtin_amdgcn_wmma_f32_16x16x32_f16(
                    false, a[ms].v, false, b, (short)0, acc[ms][t], false, false);
        }
    }

    // Quantize to 0.05 bins: f32-exact to global qsims, f16 copy to LDS
    {
        float* qbase = qsims + (size_t)rowbase * A_DIM;
#pragma unroll
        for (int ms = 0; ms < 4; ++ms) {
#pragma unroll
            for (int t = 0; t < 4; ++t) {
                const int nbase = wv * 64 + t * 16;
#pragma unroll
                for (int i = 0; i < 8; ++i) {
                    const int m = ms * 16 + khalf * 8 + i;
                    const float q = rintf(acc[ms][t][i] * INV_BIN) * BIN;
                    swbuf[m * SWS + nbase + mrow] = (_Float16)q;
                    qbase[(size_t)m * A_DIM + nbase + mrow] = q;
                }
            }
        }
    }
    __syncthreads();

    // ---------------- Stage 2: softmax, weights overwrite sims in place ----
    {
        const int row = tid >> 3;          // 0..63
        const int sub = tid & 7;
        float mx = -INFINITY;
        for (int c = sub; c < A_DIM; c += 8)
            mx = fmaxf(mx, (float)swbuf[row * SWS + c]);
        mx = fmaxf(mx, __shfl_xor(mx, 1, 32));
        mx = fmaxf(mx, __shfl_xor(mx, 2, 32));
        mx = fmaxf(mx, __shfl_xor(mx, 4, 32));
        float sum = 0.0f;
        for (int c = sub; c < A_DIM; c += 8)
            sum += __expf((float)swbuf[row * SWS + c] - mx);
        sum += __shfl_xor(sum, 1, 32);
        sum += __shfl_xor(sum, 2, 32);
        sum += __shfl_xor(sum, 4, 32);
        const float rinv = 1.0f / sum;
        // each thread re-reads then overwrites only its own elements: race-free
        for (int c = sub; c < A_DIM; c += 8) {
            const float w = __expf((float)swbuf[row * SWS + c] - mx) * rinv;
            swbuf[row * SWS + c] = (_Float16)w;
        }
    }
    __syncthreads();

    // ---------------- Stage 3: out via WMMA (M=64, N=512, K=1024) ----------
    // wave wv owns H columns [wv*32, wv*32+32) as 2 N-tiles x 4 M-subs
    v8f acc2[4][2];
#pragma unroll
    for (int ms = 0; ms < 4; ++ms)
#pragma unroll
        for (int t = 0; t < 2; ++t) acc2[ms][t] = (v8f)0.0f;

    for (int kt = 0; kt < A_DIM / 32; ++kt) {      // 32 k-steps of 32
        const int k0 = kt * 32 + khalf * 16;
        AFrag a[4];
#pragma unroll
        for (int ms = 0; ms < 4; ++ms) {
            a[ms].h8[0] = *(const v8h*)&swbuf[(ms * 16 + mrow) * SWS + k0];
            a[ms].h8[1] = *(const v8h*)&swbuf[(ms * 16 + mrow) * SWS + k0 + 8];
        }
#pragma unroll
        for (int t = 0; t < 2; ++t) {
            const int hrow = wv * 32 + t * 16 + mrow;   // output column n
            const _Float16* bp = vt + (size_t)hrow * A_DIM + kt * 32 + khalf * 16;
            const v16h b = *(const v16h*)bp;
#pragma unroll
            for (int ms = 0; ms < 4; ++ms)
                acc2[ms][t] = __builtin_amdgcn_wmma_f32_16x16x32_f16(
                    false, a[ms].v, false, b, (short)0, acc2[ms][t], false, false);
        }
    }

    float* obase = out + (size_t)rowbase * H_DIM;
#pragma unroll
    for (int ms = 0; ms < 4; ++ms) {
#pragma unroll
        for (int t = 0; t < 2; ++t) {
            const int nb = wv * 32 + t * 16;
#pragma unroll
            for (int i = 0; i < 8; ++i) {
                const int m = ms * 16 + khalf * 8 + i;
                obase[(size_t)m * H_DIM + nb + mrow] = acc2[ms][t][i];
            }
        }
    }
}

// ---------------------------------------------------------------------------
// Launch
// ---------------------------------------------------------------------------
extern "C" void kernel_launch(void* const* d_in, const int* in_sizes, int n_in,
                              void* d_out, int out_size, void* d_ws, size_t ws_size,
                              hipStream_t stream) {
    const float* x       = (const float*)d_in[0];   // [B][384]
    const float* anchors = (const float*)d_in[1];   // [1024][384]
    const float* values  = (const float*)d_in[2];   // [1024][512]

    const int B = in_sizes[0] / D_DIM;

    float* out   = (float*)d_out;                       // [B][512] first
    float* qsims = out + (size_t)B * H_DIM;             // then [B][1024]

    _Float16* ah = (_Float16*)d_ws;                     // 1024*384 f16 = 768 KB
    _Float16* vt = ah + (size_t)A_DIM * D_DIM;          // 512*1024 f16 = 1 MB

    ra_prep_anchors<<<A_DIM, 128, 0, stream>>>(anchors, ah);
    ra_prep_values<<<(A_DIM * H_DIM) / 256, 256, 0, stream>>>(values, vt);
    ra_main<<<B / TILE_M, NTHREADS, 0, stream>>>(x, ah, vt, out, qsims);
}
